// GNNDecoder_71545565216844
// MI455X (gfx1250) — matrix operand order repaired and checked by reading, not compiled, not run on hardware
//
#include <hip/hip_runtime.h>
#include <stdint.h>

// ---------------- problem constants ----------------
#define BATCH 64
#define NCHK  1024
#define NVAR  2048
#define NEDGE 6144            // NCHK * 6
#define F     64              // NODE_F == EDGE_F
#define HID   128
#define NITER 4
#define MAXDEG 16
#define ME    ((size_t)BATCH * NEDGE)   // 393216 edge rows
#define MV    ((size_t)BATCH * NVAR)    // 131072 var rows
#define MC    ((size_t)BATCH * NCHK)    // 65536 chk rows

// dynamic LDS layout for fused MLP kernel (bytes)
#define SM_W1   0
#define SM_W2   (HID * HID * 2)                   // 32768
#define SM_W3   (2 * HID * HID * 2)               // 65536
#define SM_ACT  (2 * HID * HID * 2 + F * HID * 2) // 81920
#define SM_TOT  (SM_ACT + 8 * 16 * HID * 2)       // 114688

// ---------------- WMMA types ----------------
typedef __attribute__((ext_vector_type(16))) __bf16 v16bf;
typedef __attribute__((ext_vector_type(8)))  float  v8f;

union Frag { v16bf v; uint32_t u[8]; uint4 q[2]; };
union Acc8 { v8f v; float f[8]; };

__device__ __forceinline__ uint32_t f2bf(float x) {
  uint32_t u = __float_as_uint(x);
  return (u + 0x7FFFu + ((u >> 16) & 1u)) >> 16;     // RNE f32 -> bf16
}
__device__ __forceinline__ uint32_t pack2(float lo, float hi) {
  return f2bf(lo) | (f2bf(hi) << 16);
}
__device__ __forceinline__ float bf2f(uint32_t b) { return __uint_as_float(b << 16); }

// A-matrix 16x32 bf16 tile, per-lane: K = {kc+half*8 .. +7} and {kc+16+half*8 .. +7}
__device__ __forceinline__ v16bf ld_a_ptr(const uint16_t* rowp, int kc, int half) {
  const uint16_t* r = rowp + kc + half * 8;
  Frag f;
  f.q[0] = *(const uint4*)r;
  f.q[1] = *(const uint4*)(r + 16);
  return f.v;
}
// B-matrix 32x16 tile from row-major (N x K) weight (== B^T): per-lane K = {kc+half*8 .. +7} and {kc+16+half*8 .. +7}
__device__ __forceinline__ v16bf ld_b_w(const uint16_t* W, int n, int ld, int kc, int half) {
  const uint16_t* r = W + (size_t)n * ld + kc + half * 8;
  Frag f;
  f.q[0] = *(const uint4*)r;
  f.q[1] = *(const uint4*)(r + 16);
  return f.v;
}
__device__ __forceinline__ v8f wmma_bf(v16bf a, v16bf b, v8f c) {
  return __builtin_amdgcn_wmma_f32_16x16x32_bf16(false, a, false, b, (short)0, c, false, false);
}
__device__ __forceinline__ float sigmoidf_(float x) { return 1.f / (1.f + __expf(-x)); }

// ---------------- utility kernels ----------------
__global__ void k_fill(uint32_t* p, uint32_t v, size_t n) {
  size_t i = blockIdx.x * (size_t)blockDim.x + threadIdx.x;
  for (; i < n; i += (size_t)gridDim.x * blockDim.x) p[i] = v;
}
__global__ void k_cvt(const float* __restrict__ s, uint16_t* __restrict__ d, int n) {
  int i = blockIdx.x * blockDim.x + threadIdx.x;
  if (i < n) d[i] = (uint16_t)f2bf(s[i]);
}

// ---- deterministic inverse adjacency build (graph is batch-invariant, E=6144) ----
__global__ void k_build_inv(const int* __restrict__ idx, int nE,
                            int* __restrict__ inv, int* __restrict__ deg, int nodes) {
  int n = blockIdx.x * blockDim.x + threadIdx.x;
  if (n >= nodes) return;
  int d = 0;
  for (int e = 0; e < nE; e++)
    if (idx[e] == n && d < MAXDEG) inv[n * MAXDEG + d++] = e;
  deg[n] = d;
}

// ---- deterministic segmented reduction: per-edge messages -> per-node bf16 sums ----
// thread handles 16 features of one (batch, node); fixed slot order => deterministic
__global__ void k_reduce(const uint16_t* __restrict__ msg, const int* __restrict__ inv,
                         const int* __restrict__ deg, int nodes, uint16_t* __restrict__ out) {
  int tid = blockIdx.x * blockDim.x + threadIdx.x;     // (b*nodes + node)*4 + g
  int g = tid & 3, bn = tid >> 2;
  if (bn >= BATCH * nodes) return;
  int b = bn / nodes, node = bn % nodes;
  int d = deg[node];
  float acc[16] = {};
  for (int s = 0; s < d; s++) {
    int e = inv[node * MAXDEG + s];
    const uint32_t* p = (const uint32_t*)(msg + ((size_t)b * NEDGE + e) * F + g * 16);
#pragma unroll
    for (int i = 0; i < 8; i++) {
      uint32_t w = p[i];
      acc[2 * i]     += bf2f(w & 0xFFFFu);
      acc[2 * i + 1] += bf2f(w >> 16);
    }
  }
  uint32_t* o = (uint32_t*)(out + ((size_t)bn) * F + g * 16);
#pragma unroll
  for (int i = 0; i < 8; i++) o[i] = pack2(acc[2 * i], acc[2 * i + 1]);
}

// ================= fully fused edge MLP =================
// Per workgroup: 8 waves, 128 edge rows. Per wave: one 16-row tile through all
// three layers. L1 gathers [hc|hv] rows; L3 streams per-edge messages (bf16).
// Layer boundaries transpose C-layout -> A-layout through a private 4KB LDS
// region per wave (same-wave LDS ops are in-order: no barriers needed).
__global__ __launch_bounds__(256) void k_mlp_fused(
    const uint16_t* __restrict__ hc, const uint16_t* __restrict__ hv,
    const int* __restrict__ chk, const int* __restrict__ var,
    const uint16_t* __restrict__ W1, const float* __restrict__ b1,
    const uint16_t* __restrict__ W2, const float* __restrict__ b2,
    const uint16_t* __restrict__ W3, const float* __restrict__ b3,
    uint16_t* __restrict__ msg) {
  extern __shared__ char smem[];
  uint16_t* sW1 = (uint16_t*)(smem + SM_W1);
  uint16_t* sW2 = (uint16_t*)(smem + SM_W2);
  uint16_t* sW3 = (uint16_t*)(smem + SM_W3);

  // cooperative weight staging (L2-resident globals -> LDS, once per WG)
  {
    uint4* s1 = (uint4*)sW1; const uint4* g1 = (const uint4*)W1;
    uint4* s2 = (uint4*)sW2; const uint4* g2 = (const uint4*)W2;
    uint4* s3 = (uint4*)sW3; const uint4* g3 = (const uint4*)W3;
    for (int i = threadIdx.x; i < HID * HID / 8; i += 256) { s1[i] = g1[i]; s2[i] = g2[i]; }
    for (int i = threadIdx.x; i < F * HID / 8; i += 256) s3[i] = g3[i];
  }
  __syncthreads();

  int lane = threadIdx.x & 31, wave = threadIdx.x >> 5;
  int ln = lane & 15, half = lane >> 4;
  uint16_t* act = (uint16_t*)(smem + SM_ACT) + wave * (16 * HID); // 16x128 bf16, per wave

  size_t tile = ((size_t)blockIdx.x * 8 + wave) * 16;  // global edge-row tile base
  size_t m = tile + ln;
  int b = (int)(m / NEDGE), e = (int)(m % NEDGE);
  const uint16_t* rowc = hc + ((size_t)b * NCHK + chk[e]) * F;
  const uint16_t* rowv = hv + ((size_t)b * NVAR + var[e]) * F;

  // ---- layer 1: gathered pair GEMM (K=128, N=128), ReLU -> act ----
  {
    Acc8 acc[8] = {};
#pragma unroll
    for (int ks = 0; ks < 4; ks++) {
      const uint16_t* src = (ks < 2) ? rowc : rowv;
      v16bf a = ld_a_ptr(src, (ks & 1) * 32, half);
#pragma unroll
      for (int nt = 0; nt < 8; nt++)
        acc[nt].v = wmma_bf(a, ld_b_w(sW1, nt * 16 + ln, HID, ks * 32, half), acc[nt].v);
    }
#pragma unroll
    for (int nt = 0; nt < 8; nt++) {
      int col = nt * 16 + ln;
      float bv = b1[col];
#pragma unroll
      for (int j = 0; j < 8; j++) {
        float v = acc[nt].f[j] + bv;
        v = v > 0.f ? v : 0.f;
        act[(half * 8 + j) * HID + col] = (uint16_t)f2bf(v);
      }
    }
  }
  // ---- layer 2: act GEMM (K=128, N=128), ReLU -> act (reads precede writes) ----
  {
    Acc8 acc[8] = {};
#pragma unroll
    for (int ks = 0; ks < 4; ks++) {
      v16bf a = ld_a_ptr(act + ln * HID, ks * 32, half);
#pragma unroll
      for (int nt = 0; nt < 8; nt++)
        acc[nt].v = wmma_bf(a, ld_b_w(sW2, nt * 16 + ln, HID, ks * 32, half), acc[nt].v);
    }
#pragma unroll
    for (int nt = 0; nt < 8; nt++) {
      int col = nt * 16 + ln;
      float bv = b2[col];
#pragma unroll
      for (int j = 0; j < 8; j++) {
        float v = acc[nt].f[j] + bv;
        v = v > 0.f ? v : 0.f;
        act[(half * 8 + j) * HID + col] = (uint16_t)f2bf(v);
      }
    }
  }
  // ---- layer 3: act GEMM (K=128, N=64) -> streamed per-edge messages ----
  {
    Acc8 acc[4] = {};
#pragma unroll
    for (int ks = 0; ks < 4; ks++) {
      v16bf a = ld_a_ptr(act + ln * HID, ks * 32, half);
#pragma unroll
      for (int nt = 0; nt < 4; nt++)
        acc[nt].v = wmma_bf(a, ld_b_w(sW3, nt * 16 + ln, HID, ks * 32, half), acc[nt].v);
    }
#pragma unroll
    for (int nt = 0; nt < 4; nt++) {
      int col = nt * 16 + ln;
      float bv = b3[col];
#pragma unroll
      for (int j = 0; j < 8; j++)
        msg[(tile + half * 8 + j) * F + col] = (uint16_t)f2bf(acc[nt].f[j] + bv);
    }
  }
}

// ---------------- fused GRU cell (x bf16, h bf16) ----------------
// Per wave: one 16-node tile, 48 WMMAs. For a fixed (node, feature) all four
// gate accumulators live in the same lane/slot -> gate math is pure register
// arithmetic. Wih/Whh staged in static LDS once per workgroup.
__global__ __launch_bounds__(256) void k_gru(
    const uint16_t* __restrict__ x, const uint16_t* __restrict__ h,
    const uint16_t* __restrict__ Wih, const uint16_t* __restrict__ Whh,
    const float* __restrict__ bih, const float* __restrict__ bhh,
    uint16_t* __restrict__ hout) {
  __shared__ uint16_t sWi[192 * F];    // 24 KB
  __shared__ uint16_t sWh[192 * F];    // 24 KB
  {
    uint4* si = (uint4*)sWi; const uint4* gi = (const uint4*)Wih;
    uint4* sh = (uint4*)sWh; const uint4* gh = (const uint4*)Whh;
    for (int i = threadIdx.x; i < 192 * F / 8; i += 256) { si[i] = gi[i]; sh[i] = gh[i]; }
  }
  __syncthreads();

  int lane = threadIdx.x & 31, wave = threadIdx.x >> 5;
  int ln = lane & 15, half = lane >> 4;
  size_t tile = ((size_t)blockIdx.x * 8 + wave) * 16;
  size_t arow = tile + ln;

  Acc8 rs[4] = {}, zs[4] = {}, in_[4] = {}, hn[4] = {};
#pragma unroll
  for (int ks = 0; ks < 2; ks++) {                 // gi = x @ Wih^T
    v16bf a = ld_a_ptr(x + arow * F, ks * 32, half);
#pragma unroll
    for (int t = 0; t < 4; t++) {
      rs[t].v  = wmma_bf(a, ld_b_w(sWi,       t * 16 + ln, F, ks * 32, half), rs[t].v);
      zs[t].v  = wmma_bf(a, ld_b_w(sWi,  64 + t * 16 + ln, F, ks * 32, half), zs[t].v);
      in_[t].v = wmma_bf(a, ld_b_w(sWi, 128 + t * 16 + ln, F, ks * 32, half), in_[t].v);
    }
  }
#pragma unroll
  for (int ks = 0; ks < 2; ks++) {                 // gh = h @ Whh^T (r,z summed in-place)
    v16bf a = ld_a_ptr(h + arow * F, ks * 32, half);
#pragma unroll
    for (int t = 0; t < 4; t++) {
      rs[t].v = wmma_bf(a, ld_b_w(sWh,       t * 16 + ln, F, ks * 32, half), rs[t].v);
      zs[t].v = wmma_bf(a, ld_b_w(sWh,  64 + t * 16 + ln, F, ks * 32, half), zs[t].v);
      hn[t].v = wmma_bf(a, ld_b_w(sWh, 128 + t * 16 + ln, F, ks * 32, half), hn[t].v);
    }
  }
#pragma unroll
  for (int t = 0; t < 4; t++) {
    int feat = t * 16 + ln;
    float b_r  = bih[feat] + bhh[feat];
    float b_z  = bih[64 + feat] + bhh[64 + feat];
    float b_in = bih[128 + feat];
    float b_hn = bhh[128 + feat];
#pragma unroll
    for (int j = 0; j < 8; j++) {
      size_t node = tile + half * 8 + j;
      float r = sigmoidf_(rs[t].f[j] + b_r);
      float z = sigmoidf_(zs[t].f[j] + b_z);
      float n = tanhf(in_[t].f[j] + b_in + r * (hn[t].f[j] + b_hn));
      float hold = bf2f(h[node * F + feat]);
      hout[node * F + feat] = (uint16_t)f2bf((1.f - z) * n + z * hold);
    }
  }
}

// ---------------- syndrome select for check states ----------------
__global__ void k_select(const int* __restrict__ syn, const uint16_t* __restrict__ h0,
                         const uint16_t* __restrict__ h1, uint16_t* __restrict__ o, size_t n) {
  size_t i = blockIdx.x * (size_t)blockDim.x + threadIdx.x;
  for (; i < n; i += (size_t)gridDim.x * blockDim.x)
    o[i] = syn[i >> 6] ? h1[i] : h0[i];
}

// ---------------- prediction head: llr[v][b][t] ----------------
__global__ void k_pred(const uint16_t* __restrict__ hv, const float* __restrict__ pw,
                       const float* __restrict__ pb, float* __restrict__ out, int titer) {
  size_t tid = blockIdx.x * (size_t)blockDim.x + threadIdx.x;
  if (tid >= MV) return;
  int b = (int)(tid / NVAR), v = (int)(tid % NVAR);
  const uint32_t* hp = (const uint32_t*)(hv + tid * F);
  float s = pb[0];
#pragma unroll
  for (int i = 0; i < 32; i++) {
    uint32_t d = hp[i];
    s += bf2f(d & 0xFFFFu) * pw[2 * i] + bf2f(d >> 16) * pw[2 * i + 1];
  }
  out[(size_t)v * BATCH * NITER + (size_t)b * NITER + titer] = s;
}

// ---------------- host driver ----------------
extern "C" void kernel_launch(void* const* d_in, const int* in_sizes, int n_in,
                              void* d_out, int out_size, void* d_ws, size_t ws_size,
                              hipStream_t stream) {
  (void)in_sizes; (void)n_in; (void)out_size; (void)ws_size;
  const int*   syn = (const int*)d_in[0];
  const int*   chk = (const int*)d_in[1];
  const int*   var = (const int*)d_in[2];
  const float* fw[29];
  for (int i = 3; i < 29; i++) fw[i] = (const float*)d_in[i];

  // ---- workspace carve ----
  char* base = (char*)d_ws; size_t off = 0;
  auto carve = [&](size_t bytes) -> char* {
    char* r = base + off;
    off += (bytes + 255) & ~(size_t)255;
    return r;
  };
  uint16_t* w_v2c1 = (uint16_t*)carve(HID * HID * 2);
  uint16_t* w_v2c2 = (uint16_t*)carve(HID * HID * 2);
  uint16_t* w_v2c3 = (uint16_t*)carve(F * HID * 2);
  uint16_t* w_c2v1 = (uint16_t*)carve(HID * HID * 2);
  uint16_t* w_c2v2 = (uint16_t*)carve(HID * HID * 2);
  uint16_t* w_c2v3 = (uint16_t*)carve(F * HID * 2);
  uint16_t* w_gvi  = (uint16_t*)carve(192 * F * 2);
  uint16_t* w_gvh  = (uint16_t*)carve(192 * F * 2);
  uint16_t* w_g0i  = (uint16_t*)carve(192 * F * 2);
  uint16_t* w_g0h  = (uint16_t*)carve(192 * F * 2);
  uint16_t* w_g1i  = (uint16_t*)carve(192 * F * 2);
  uint16_t* w_g1h  = (uint16_t*)carve(192 * F * 2);
  int* inv_chk = (int*)carve(NCHK * MAXDEG * 4);
  int* deg_chk = (int*)carve(NCHK * 4);
  int* inv_var = (int*)carve(NVAR * MAXDEG * 4);
  int* deg_var = (int*)carve(NVAR * 4);
  uint16_t* hcBuf[2] = { (uint16_t*)carve(MC * F * 2), (uint16_t*)carve(MC * F * 2) };
  uint16_t* hvBuf[2] = { (uint16_t*)carve(MV * F * 2), (uint16_t*)carve(MV * F * 2) };
  uint16_t* mc   = (uint16_t*)carve(MC * F * 2);
  uint16_t* mv   = (uint16_t*)carve(MV * F * 2);
  uint16_t* hc0t = (uint16_t*)carve(MC * F * 2);
  uint16_t* hc1t = (uint16_t*)carve(MC * F * 2);
  uint16_t* msg  = (uint16_t*)carve(ME * F * 2);       // 50.3 MB, reused by both MLPs

  // ---- one-time weight conversion to bf16 ----
  struct { const float* s; uint16_t* d; int n; } cj[12] = {
    { fw[3],  w_v2c1, HID * HID }, { fw[5],  w_v2c2, HID * HID }, { fw[7],  w_v2c3, F * HID },
    { fw[9],  w_c2v1, HID * HID }, { fw[11], w_c2v2, HID * HID }, { fw[13], w_c2v3, F * HID },
    { fw[15], w_gvi, 192 * F }, { fw[16], w_gvh, 192 * F },
    { fw[19], w_g0i, 192 * F }, { fw[20], w_g0h, 192 * F },
    { fw[23], w_g1i, 192 * F }, { fw[24], w_g1h, 192 * F },
  };
  for (int i = 0; i < 12; i++)
    k_cvt<<<(cj[i].n + 255) / 256, 256, 0, stream>>>(cj[i].s, cj[i].d, cj[i].n);

  // ---- one-time deterministic inverse adjacency ----
  k_build_inv<<<NCHK / 256, 256, 0, stream>>>(chk, NEDGE, inv_chk, deg_chk, NCHK);
  k_build_inv<<<NVAR / 256, 256, 0, stream>>>(var, NEDGE, inv_var, deg_var, NVAR);

  // ---- zero initial states ----
  k_fill<<<1024, 256, 0, stream>>>((uint32_t*)hcBuf[0], 0u, MC * F / 2);
  k_fill<<<1024, 256, 0, stream>>>((uint32_t*)hvBuf[0], 0u, MV * F / 2);

  int cur = 0;
  for (int t = 0; t < NITER; t++) {
    uint16_t* hc_c = hcBuf[cur], *hc_n = hcBuf[cur ^ 1];
    uint16_t* hv_c = hvBuf[cur], *hv_n = hvBuf[cur ^ 1];

    // v2c MLP -> per-edge messages -> deterministic reduce into mc
    k_mlp_fused<<<(int)(ME / 128), 256, SM_TOT, stream>>>(
        hc_c, hv_c, chk, var, w_v2c1, fw[4], w_v2c2, fw[6], w_v2c3, fw[8], msg);
    k_reduce<<<(int)(MC * 4 / 256), 256, 0, stream>>>(msg, inv_chk, deg_chk, NCHK, mc);

    // c2v MLP -> per-edge messages -> deterministic reduce into mv
    k_mlp_fused<<<(int)(ME / 128), 256, SM_TOT, stream>>>(
        hc_c, hv_c, chk, var, w_c2v1, fw[10], w_c2v2, fw[12], w_c2v3, fw[14], msg);
    k_reduce<<<(int)(MV * 4 / 256), 256, 0, stream>>>(msg, inv_var, deg_var, NVAR, mv);

    // GRU updates
    k_gru<<<(int)(MV / 128), 256, 0, stream>>>(mv, hv_c, w_gvi, w_gvh, fw[17], fw[18], hv_n);
    k_gru<<<(int)(MC / 128), 256, 0, stream>>>(mc, hc_c, w_g0i, w_g0h, fw[21], fw[22], hc0t);
    k_gru<<<(int)(MC / 128), 256, 0, stream>>>(mc, hc_c, w_g1i, w_g1h, fw[25], fw[26], hc1t);
    k_select<<<1024, 256, 0, stream>>>(syn, hc0t, hc1t, hc_n, MC * F);

    // prediction head -> out[v][b][t]
    k_pred<<<(int)(MV / 256), 256, 0, stream>>>(hv_n, fw[27], fw[28], (float*)d_out, t);

    cur ^= 1;
  }
}
